// CrossAttentionLayer_3478923510516
// MI455X (gfx1250) — matrix operand order repaired and checked by reading, not compiled, not run on hardware
//
#include <hip/hip_runtime.h>
#include <hip/hip_bf16.h>

typedef __attribute__((ext_vector_type(16))) _Float16 v16h;
typedef __attribute__((ext_vector_type(8)))  _Float16 v8h;
typedef __attribute__((ext_vector_type(4)))  _Float16 v4h;
typedef __attribute__((ext_vector_type(8)))  float    v8f;

#define D_MODEL 1024
#define NHEAD   16
#define HDIM    64
#define SEQ     512
#define BATCH   16
#define MTOT    (BATCH * SEQ)          // 8192 rows
#define SCALE   0.125f                 // 64^-0.5
#define HALFWIN 64
#define LDS_STRIDE 40                  // halves; 80 B rows keep b128 alignment

// ---------------------------------------------------------------- helpers
__device__ __forceinline__ v16h frag16(const _Float16* lo, const _Float16* hi) {
    v8h a = *(const v8h*)lo;
    v8h b = *(const v8h*)hi;
    v16h r;
#pragma unroll
    for (int i = 0; i < 8; ++i) { r[i] = a[i]; r[i + 8] = b[i]; }
    return r;
}

__device__ __forceinline__ v16h zero16h() {
    v16h r;
#pragma unroll
    for (int i = 0; i < 16; ++i) r[i] = (_Float16)0.0f;
    return r;
}

__device__ __forceinline__ float rowmax16(float v) {
    v = fmaxf(v, __shfl_xor(v, 1, 16));
    v = fmaxf(v, __shfl_xor(v, 2, 16));
    v = fmaxf(v, __shfl_xor(v, 4, 16));
    v = fmaxf(v, __shfl_xor(v, 8, 16));
    return v;
}

__device__ __forceinline__ float rowsum16(float v) {
    v += __shfl_xor(v, 1, 16);
    v += __shfl_xor(v, 2, 16);
    v += __shfl_xor(v, 4, 16);
    v += __shfl_xor(v, 8, 16);
    return v;
}

// CDNA5 async global -> LDS copy (ASYNCcnt tracked), 16 bytes per lane.
__device__ __forceinline__ void async_copy16(const _Float16* g, _Float16* l) {
    unsigned lo = (unsigned)(unsigned long long)(uintptr_t)l;  // low 32 bits used as LDS address
    unsigned long long ga = (unsigned long long)(uintptr_t)g;
    asm volatile("global_load_async_to_lds_b128 %0, %1, off"
                 :: "v"(lo), "v"(ga)
                 : "memory");
}
__device__ __forceinline__ void wait_async0() {
    asm volatile("s_wait_asynccnt 0" ::: "memory");
}

// ---------------------------------------------------------------- cast f32 -> f16
__global__ __launch_bounds__(256) void cast_f32_f16(const float* __restrict__ in,
                                                    _Float16* __restrict__ out, int n4) {
    int i = blockIdx.x * 256 + threadIdx.x;
    if (i < n4) {
        float4 v = ((const float4*)in)[i];
        v4h o;
        o[0] = (_Float16)v.x; o[1] = (_Float16)v.y;
        o[2] = (_Float16)v.z; o[3] = (_Float16)v.w;
        ((v4h*)out)[i] = o;
    }
}

// ---------------------------------------------------------------- WMMA GEMM
// C[row, col] = sum_k A[row,k] * W[col,k] + bias[col]
// A: f16 [M x 1024] row-major.  W: f16 [1024 x 1024] row-major (out x in).
// WG tile 128x128, 4 waves of 64x64, K-step 32, double-buffered async staging.
// mode 0: f16 out, head layout   q[((b*16+h)*512 + s)*64 + d]
// mode 1: f16 out, head-T layout vt[((b*16+h)*64 + d)*512 + s]
// mode 2: f32 out, sigmoid(val), row-major [row*1024 + col]
// mode 3: f32 out, raw val,      row-major [row*1024 + col]
__global__ __launch_bounds__(128) void gemm_wmma(const _Float16* __restrict__ A,
                                                 const _Float16* __restrict__ W,
                                                 const float* __restrict__ bias,
                                                 void* __restrict__ out, int mode) {
    __shared__ _Float16 ldsA[2][128 * LDS_STRIDE];
    __shared__ _Float16 ldsB[2][128 * LDS_STRIDE];

    const int tid   = threadIdx.x;
    const int wave  = tid >> 5;
    const int lane  = tid & 31;
    const int hf    = lane >> 4;        // which 16-lane half
    const int lrow  = lane & 15;
    const int waveM = wave >> 1;        // 0..1
    const int waveN = wave & 1;         // 0..1
    const int colBase = blockIdx.x * 128;
    const int rowBase = blockIdx.y * 128;

    v8f acc[4][4];
#pragma unroll
    for (int mi = 0; mi < 4; ++mi)
#pragma unroll
        for (int ni = 0; ni < 4; ++ni)
#pragma unroll
            for (int r = 0; r < 8; ++r) acc[mi][ni][r] = 0.0f;

    const _Float16* srcA = A + (size_t)(rowBase + tid) * D_MODEL;  // one A row / thread
    const _Float16* srcB = W + (size_t)(colBase + tid) * D_MODEL;  // one W row / thread

    // issue async staging of one 128x32 K-slab (A and B) into buffer `buf`
    auto stage = [&](int k0, int buf) {
        _Float16* dA = &ldsA[buf][tid * LDS_STRIDE];
        _Float16* dB = &ldsB[buf][tid * LDS_STRIDE];
#pragma unroll
        for (int c = 0; c < 4; ++c) {
            async_copy16(srcA + k0 + c * 8, dA + c * 8);
            async_copy16(srcB + k0 + c * 8, dB + c * 8);
        }
    };

    stage(0, 0);

    const int NK = D_MODEL / 32;  // 32 K-steps
    for (int kt = 0; kt < NK; ++kt) {
        const int buf = kt & 1;
        wait_async0();        // my async copies (into `buf`) done
        __syncthreads();      // everyone's copies done; prior reads of other buf done
        if (kt + 1 < NK) stage((kt + 1) * 32, buf ^ 1);

        v16h afr[4], bfr[4];
#pragma unroll
        for (int mi = 0; mi < 4; ++mi) {
            const _Float16* p =
                &ldsA[buf][(waveM * 64 + mi * 16 + lrow) * LDS_STRIDE + 8 * hf];
            afr[mi] = frag16(p, p + 16);
        }
#pragma unroll
        for (int ni = 0; ni < 4; ++ni) {
            const _Float16* p =
                &ldsB[buf][(waveN * 64 + ni * 16 + lrow) * LDS_STRIDE + 8 * hf];
            bfr[ni] = frag16(p, p + 16);
        }
#pragma unroll
        for (int mi = 0; mi < 4; ++mi)
#pragma unroll
            for (int ni = 0; ni < 4; ++ni)
                acc[mi][ni] = __builtin_amdgcn_wmma_f32_16x16x32_f16(
                    false, afr[mi], false, bfr[ni], (short)0, acc[mi][ni], false, false);
    }

    // epilogue
#pragma unroll
    for (int ni = 0; ni < 4; ++ni) {
        const int col = colBase + waveN * 64 + ni * 16 + lrow;
        const float bv = bias[col];
        const int h = col >> 6, d = col & 63;
#pragma unroll
        for (int mi = 0; mi < 4; ++mi) {
#pragma unroll
            for (int r = 0; r < 8; ++r) {
                const int rowg = rowBase + waveM * 64 + mi * 16 + r + 8 * hf;
                const float val = acc[mi][ni][r] + bv;
                if (mode == 0) {
                    const int b = rowg >> 9, s = rowg & 511;
                    ((_Float16*)out)[(((size_t)(b * NHEAD + h) * SEQ) + s) * HDIM + d] =
                        (_Float16)val;
                } else if (mode == 1) {
                    const int b = rowg >> 9, s = rowg & 511;
                    ((_Float16*)out)[(((size_t)(b * NHEAD + h) * HDIM) + d) * SEQ + s] =
                        (_Float16)val;
                } else if (mode == 2) {
                    ((float*)out)[(size_t)rowg * D_MODEL + col] =
                        1.0f / (1.0f + __expf(-val));
                } else {
                    ((float*)out)[(size_t)rowg * D_MODEL + col] = val;
                }
            }
        }
    }
}

// ---------------------------------------------------------------- windowed flash attention
// Q,K: [bh][s][d] f16 ; Vt: [bh][d][s] f16 ; attn out: [b*512+s][1024] f16
__global__ __launch_bounds__(128) void attn_wmma(const _Float16* __restrict__ Q,
                                                 const _Float16* __restrict__ K,
                                                 const _Float16* __restrict__ Vt,
                                                 _Float16* __restrict__ attn) {
    __shared__ _Float16 ldsP[4][16 * LDS_STRIDE];

    const int wave = threadIdx.x >> 5;
    const int lane = threadIdx.x & 31;
    const int hf   = lane >> 4;
    const int lrow = lane & 15;

    const int qt = blockIdx.x * 4 + wave;     // 0..8191
    const int bh = qt >> 5;                   // (b*16 + h)
    const int q0 = (qt & 31) << 4;

    const _Float16* Qb = Q  + (size_t)bh * SEQ * HDIM;
    const _Float16* Kb = K  + (size_t)bh * SEQ * HDIM;
    const _Float16* Vb = Vt + (size_t)bh * HDIM * SEQ;

    // Q fragments (persist): 16 x 64 as two 16x32 A tiles
    v16h aq[2];
#pragma unroll
    for (int kk = 0; kk < 2; ++kk) {
        const _Float16* p = Qb + (size_t)(q0 + lrow) * HDIM + 32 * kk + 8 * hf;
        aq[kk] = frag16(p, p + 16);
    }

    v8f o[4];
#pragma unroll
    for (int dt = 0; dt < 4; ++dt)
#pragma unroll
        for (int r = 0; r < 8; ++r) o[dt][r] = 0.0f;
    float m[8], l[8];
#pragma unroll
    for (int r = 0; r < 8; ++r) { m[r] = -3.0e38f; l[r] = 0.0f; }

    for (int it = 0; it < 5; ++it) {
        const int j0 = q0 - HALFWIN + it * 32;

        // scores for two 16-key tiles
        v8f sc[2];
#pragma unroll
        for (int jt = 0; jt < 2; ++jt) {
            const int key = j0 + jt * 16 + lrow;
            const bool kvalid = (key >= 0) && (key < SEQ);
            v8f s;
#pragma unroll
            for (int r = 0; r < 8; ++r) s[r] = 0.0f;
#pragma unroll
            for (int kk = 0; kk < 2; ++kk) {
                v16h bk;
                if (kvalid) {
                    const _Float16* p = Kb + (size_t)key * HDIM + 32 * kk + 8 * hf;
                    bk = frag16(p, p + 16);
                } else {
                    bk = zero16h();
                }
                s = __builtin_amdgcn_wmma_f32_16x16x32_f16(
                    false, aq[kk], false, bk, (short)0, s, false, false);
            }
            sc[jt] = s;
        }

        // online softmax update
        float p0[8], p1[8], sf[8];
#pragma unroll
        for (int r = 0; r < 8; ++r) {
            const int qi = q0 + r + 8 * hf;
            const int k0c = j0 + lrow;
            const int k1c = j0 + 16 + lrow;
            const int d0 = qi - k0c, d1 = qi - k1c;
            const bool v0 = (k0c >= 0) && (k0c < SEQ) && (d0 <= HALFWIN) && (d0 >= -HALFWIN);
            const bool v1 = (k1c >= 0) && (k1c < SEQ) && (d1 <= HALFWIN) && (d1 >= -HALFWIN);
            const float s0v = v0 ? sc[0][r] * SCALE : -3.0e38f;
            const float s1v = v1 ? sc[1][r] * SCALE : -3.0e38f;
            const float mx = rowmax16(fmaxf(s0v, s1v));
            const float nm = fmaxf(m[r], mx);
            sf[r] = __expf(m[r] - nm);
            m[r] = nm;
            const float e0 = v0 ? __expf(s0v - nm) : 0.0f;
            const float e1 = v1 ? __expf(s1v - nm) : 0.0f;
            l[r] = l[r] * sf[r] + rowsum16(e0 + e1);
            p0[r] = e0; p1[r] = e1;
        }
#pragma unroll
        for (int dt = 0; dt < 4; ++dt)
#pragma unroll
            for (int r = 0; r < 8; ++r) o[dt][r] *= sf[r];

        // D-layout P -> LDS -> A-layout fragment (16 x 32)
        _Float16* pl = ldsP[wave];
#pragma unroll
        for (int r = 0; r < 8; ++r) {
            const int row = r + 8 * hf;
            pl[row * LDS_STRIDE + lrow]      = (_Float16)p0[r];
            pl[row * LDS_STRIDE + 16 + lrow] = (_Float16)p1[r];
        }
        v16h ap;
        {
            const _Float16* p = pl + lrow * LDS_STRIDE + 16 * hf;
            ap = frag16(p, p + 8);
        }

        // P (16x32) x V (32x64) accumulated into O
        const int kbase = j0 + 16 * hf;
        const bool vvalid = (kbase >= 0) && (kbase < SEQ);
#pragma unroll
        for (int dt = 0; dt < 4; ++dt) {
            v16h bv;
            if (vvalid) {
                const _Float16* p = Vb + (size_t)(dt * 16 + lrow) * SEQ + kbase;
                bv = frag16(p, p + 8);
            } else {
                bv = zero16h();
            }
            o[dt] = __builtin_amdgcn_wmma_f32_16x16x32_f16(
                false, ap, false, bv, (short)0, o[dt], false, false);
        }
    }

    // write merged-head attention, f16
    const int b = bh >> 4, h = bh & 15;
#pragma unroll
    for (int r = 0; r < 8; ++r) {
        const float inv = 1.0f / l[r];
        const int s = q0 + r + 8 * hf;
#pragma unroll
        for (int dt = 0; dt < 4; ++dt) {
            attn[((size_t)(b * SEQ + s)) * D_MODEL + h * HDIM + dt * 16 + lrow] =
                (_Float16)(o[dt][r] * inv);
        }
    }
}

// ---------------------------------------------------------------- fused gate/blend/LayerNorm
__global__ __launch_bounds__(256) void ln_kernel(const float* __restrict__ hidden,
                                                 const float* __restrict__ gate,
                                                 const float* __restrict__ y,
                                                 const float* __restrict__ gamma,
                                                 const float* __restrict__ beta,
                                                 float* __restrict__ out) {
    const int wave = threadIdx.x >> 5;
    const int lane = threadIdx.x & 31;
    const int row = blockIdx.x * 8 + wave;

    const float* hr = hidden + (size_t)row * D_MODEL;
    const float* gr = gate   + (size_t)row * D_MODEL;
    const float* yr = y      + (size_t)row * D_MODEL;

    float vals[32];
    float s = 0.0f, s2 = 0.0f;
#pragma unroll
    for (int i = 0; i < 32; ++i) {
        const int c = i * 32 + lane;
        const float v = 0.5f * hr[c] + 0.5f * (gr[c] * yr[c]);
        vals[i] = v;
        s += v; s2 += v * v;
    }
#pragma unroll
    for (int msk = 1; msk < 32; msk <<= 1) {
        s  += __shfl_xor(s,  msk, 32);
        s2 += __shfl_xor(s2, msk, 32);
    }
    const float mean = s * (1.0f / D_MODEL);
    const float var  = s2 * (1.0f / D_MODEL) - mean * mean;
    const float rstd = rsqrtf(var + 1e-5f);
#pragma unroll
    for (int i = 0; i < 32; ++i) {
        const int c = i * 32 + lane;
        out[(size_t)row * D_MODEL + c] = (vals[i] - mean) * rstd * gamma[c] + beta[c];
    }
}

// ---------------------------------------------------------------- launch
extern "C" void kernel_launch(void* const* d_in, const int* in_sizes, int n_in,
                              void* d_out, int out_size, void* d_ws, size_t ws_size,
                              hipStream_t stream) {
    const float* hidden = (const float*)d_in[0];
    const float* cross  = (const float*)d_in[1];
    const float* Wq = (const float*)d_in[2];  const float* bq = (const float*)d_in[3];
    const float* Wk = (const float*)d_in[4];  const float* bk = (const float*)d_in[5];
    const float* Wv = (const float*)d_in[6];  const float* bv = (const float*)d_in[7];
    const float* Wo = (const float*)d_in[8];  const float* bo = (const float*)d_in[9];
    const float* Wg = (const float*)d_in[10]; const float* bg = (const float*)d_in[11];
    const float* gamma = (const float*)d_in[12];
    const float* beta  = (const float*)d_in[13];

    char* ws = (char*)d_ws;
    const size_t MB = 1u << 20;
    _Float16* h16  = (_Float16*)(ws + 0 * MB);
    _Float16* c16  = (_Float16*)(ws + 16 * MB);
    _Float16* q16  = (_Float16*)(ws + 32 * MB);
    _Float16* k16  = (_Float16*)(ws + 48 * MB);
    _Float16* vt16 = (_Float16*)(ws + 64 * MB);
    _Float16* a16  = (_Float16*)(ws + 80 * MB);
    _Float16* wq16 = (_Float16*)(ws + 96 * MB);
    _Float16* wk16 = (_Float16*)(ws + 98 * MB);
    _Float16* wv16 = (_Float16*)(ws + 100 * MB);
    _Float16* wo16 = (_Float16*)(ws + 102 * MB);
    _Float16* wg16 = (_Float16*)(ws + 104 * MB);
    float*    gatef = (float*)(ws + 112 * MB);
    float*    yf    = (float*)(ws + 144 * MB);

    const int nAct4 = (MTOT * D_MODEL) / 4;      // 2,097,152
    const int nW4   = (D_MODEL * D_MODEL) / 4;   //   262,144
    dim3 cb(256);
    cast_f32_f16<<<dim3((nAct4 + 255) / 256), cb, 0, stream>>>(hidden, h16, nAct4);
    cast_f32_f16<<<dim3((nAct4 + 255) / 256), cb, 0, stream>>>(cross,  c16, nAct4);
    cast_f32_f16<<<dim3((nW4 + 255) / 256), cb, 0, stream>>>(Wq, wq16, nW4);
    cast_f32_f16<<<dim3((nW4 + 255) / 256), cb, 0, stream>>>(Wk, wk16, nW4);
    cast_f32_f16<<<dim3((nW4 + 255) / 256), cb, 0, stream>>>(Wv, wv16, nW4);
    cast_f32_f16<<<dim3((nW4 + 255) / 256), cb, 0, stream>>>(Wo, wo16, nW4);
    cast_f32_f16<<<dim3((nW4 + 255) / 256), cb, 0, stream>>>(Wg, wg16, nW4);

    dim3 ggrid(D_MODEL / 128, MTOT / 128);   // (8, 64)
    dim3 gblk(128);
    gemm_wmma<<<ggrid, gblk, 0, stream>>>(h16, wq16, bq, (void*)q16,  0);
    gemm_wmma<<<ggrid, gblk, 0, stream>>>(c16, wk16, bk, (void*)k16,  0);
    gemm_wmma<<<ggrid, gblk, 0, stream>>>(c16, wv16, bv, (void*)vt16, 1);
    gemm_wmma<<<ggrid, gblk, 0, stream>>>(h16, wg16, bg, (void*)gatef, 2);

    attn_wmma<<<dim3(2048), dim3(128), 0, stream>>>(q16, k16, vt16, a16);

    gemm_wmma<<<ggrid, gblk, 0, stream>>>(a16, wo16, bo, (void*)yf, 3);

    ln_kernel<<<dim3(MTOT / 8), dim3(256), 0, stream>>>(hidden, gatef, yf, gamma, beta,
                                                        (float*)d_out);
}
